// AttentionLayer_4982162063439
// MI455X (gfx1250) — matrix operand order repaired and checked by reading, not compiled, not run on hardware
//
#include <hip/hip_runtime.h>

typedef __attribute__((ext_vector_type(16))) _Float16 v16h;
typedef __attribute__((ext_vector_type(8)))  _Float16 v8h;
typedef __attribute__((ext_vector_type(2)))  _Float16 v2h;
typedef __attribute__((ext_vector_type(8)))  float    v8f;
typedef __attribute__((ext_vector_type(4)))  unsigned int u32x4;
typedef __attribute__((ext_vector_type(8)))  int          i32x8;
typedef __attribute__((ext_vector_type(4)))  int          i32x4;

#define F_DIM   256     // input feature dim (GEMM K)
#define A_DIM   128     // dense units (GEMM N)
#define KNB     32      // neighbors per node
#define NNODES  20000
#define TILE_ROWS 64
#define TILE_K    32
#define LDS_ROW   36    // 32 data dwords + 4 pad dwords (TDM pad) -> conflict-free frag reads

// ---------------------------------------------------------------------------
// Kernel 0: transpose W (F x A, f32) -> Wt (A x F, f16) and zero stats accums.
// ---------------------------------------------------------------------------
__global__ void prep_kernel(const float* __restrict__ W, _Float16* __restrict__ Wt,
                            float* __restrict__ statsS, float* __restrict__ statsN) {
  int idx = blockIdx.x * 256 + threadIdx.x;   // 0 .. 32767
  int c = idx >> 8;                           // output column 0..127
  int k = idx & 255;                          // K index 0..255
  Wt[(size_t)c * F_DIM + k] = (_Float16)W[(size_t)k * A_DIM + c];
  if (blockIdx.x == 0) {                      // 256 threads zero 2*(sum|sumsq)
    statsS[threadIdx.x] = 0.f;
    statsN[threadIdx.x] = 0.f;
  }
}

// ---------------------------------------------------------------------------
// TDM: DMA a 64-row x 32-col f32 tile (row stride 256) global -> LDS.
// D# per CDNA5 ISA. 2D tile, data_size=4B,
// pad_enable: +4 dwords every 32 dwords -> LDS row stride 36 dwords.
// tensor_dim1 = rows remaining so OOB tail rows read as zeros.
// ---------------------------------------------------------------------------
__device__ __forceinline__ void tdm_load_tile(unsigned lds_off, const float* gptr,
                                              int rows_remaining) {
  unsigned long long ga = (unsigned long long)(size_t)gptr;
  u32x4 g0;
  g0[0] = 1u;                                             // count=1 (user mode)
  g0[1] = lds_off;                                        // lds_addr (bytes)
  g0[2] = (unsigned)(ga & 0xffffffffull);                 // global_addr[31:0]
  g0[3] = (unsigned)((ga >> 32) & 0x1ffffffull)           // global_addr[56:32]
        | (2u << 30);                                     // type = 2 ("image")
  unsigned td0 = F_DIM;                                   // tensor_dim0
  unsigned td1 = (unsigned)(rows_remaining > 0 ? rows_remaining : 0);
  i32x8 g1;
  g1[0] = (int)((2u << 16)        // data_size = 4 bytes
              | (1u << 20)        // pad_enable
              | (4u << 22)        // pad_interval: 2^(4+1) = 32 dwords
              | (3u << 25));      // pad_amount:  3+1   =  4 dwords
  g1[1] = (int)((td0 & 0xffffu) << 16);                   // [63:48] td0 lo16
  g1[2] = (int)((td0 >> 16) | ((td1 & 0xffffu) << 16));   // td0 hi16 | td1 lo16
  g1[3] = (int)((td1 >> 16) | ((unsigned)TILE_K << 16));  // td1 hi16 | tile_dim0=32
  g1[4] = (int)TILE_ROWS;                                 // tile_dim1=64, tile_dim2=0
  g1[5] = (int)F_DIM;                                     // tensor_dim0_stride lo32
  g1[6] = 0;                                              // stride0 hi16 | stride1 lo16
  g1[7] = 0;                                              // stride1 hi32
  i32x4 z4; z4[0] = 0; z4[1] = 0; z4[2] = 0; z4[3] = 0;   // VADDR2/3 groups unused (2D)
  i32x8 z8;
#pragma unroll
  for (int q = 0; q < 8; ++q) z8[q] = 0;
  __builtin_amdgcn_tensor_load_to_lds(g0, g1, z4, z4, z8, 0);
}

// ---------------------------------------------------------------------------
// Kernel 1: column stats of R = X @ W + b  (R never stored).
// Block = 128 threads (4 waves). Block tile = 64 rows x 128 cols.
// TDM double-buffers X tiles into LDS; WMMA f16 does the GEMM.
// ---------------------------------------------------------------------------
__global__ void __launch_bounds__(128)
gemm_stats_kernel(const float* __restrict__ X, const _Float16* __restrict__ Wt,
                  const float* __restrict__ bias,
                  float* __restrict__ colsum, float* __restrict__ colsq, int M) {
  __shared__ float sXf[2][TILE_ROWS * LDS_ROW];
  __shared__ float sSum[A_DIM];
  __shared__ float sSq[A_DIM];

  const int tid  = threadIdx.x;
  const int wave = tid >> 5;
  const int lane = tid & 31;
  const int g    = lane >> 4;      // half-wave select (K-split of fragments)
  const int hn   = lane & 15;      // M index for A-frag, N index for B/C-frag
  const int rowBase = blockIdx.x * TILE_ROWS;
  const int rowsRem = M - rowBase;

  if (tid < A_DIM) { sSum[tid] = 0.f; sSq[tid] = 0.f; }

  v8f acc[8];
#pragma unroll
  for (int j = 0; j < 8; ++j)
#pragma unroll
    for (int r = 0; r < 8; ++r) acc[j][r] = 0.f;

  const unsigned ldsBase[2] = { (unsigned)(size_t)&sXf[0][0],
                                (unsigned)(size_t)&sXf[1][0] };
  const float* gTile = X + (size_t)rowBase * F_DIM;

  if (wave == 0) tdm_load_tile(ldsBase[0], gTile, rowsRem);

  for (int i = 0; i < 8; ++i) {
    const int buf = i & 1;
    if (wave == 0) {
      if (i < 7) {
        tdm_load_tile(ldsBase[buf ^ 1], gTile + (i + 1) * TILE_K, rowsRem);
        __builtin_amdgcn_s_wait_tensorcnt(1);   // oldest (current tile) complete
      } else {
        __builtin_amdgcn_s_wait_tensorcnt(0);
      }
    }
    __syncthreads();                            // tile visible to all waves

    // A fragment: 16x32 f16 built from f32 LDS tile (packed RTZ converts).
    //   halfs 0..15 -> K = 16g + 0..15
    const float* rowp = &sXf[buf][(wave * 16 + hn) * LDS_ROW];
    float4 f0 = *(const float4*)(rowp + 16 * g);
    float4 f1 = *(const float4*)(rowp + 16 * g + 4);
    float4 f2 = *(const float4*)(rowp + 16 * g + 8);
    float4 f3 = *(const float4*)(rowp + 16 * g + 12);
    v16h a;
    {
      v2h p;
      p = __builtin_bit_cast(v2h, __builtin_amdgcn_cvt_pkrtz(f0.x, f0.y));
      a[0]  = p[0]; a[1]  = p[1];
      p = __builtin_bit_cast(v2h, __builtin_amdgcn_cvt_pkrtz(f0.z, f0.w));
      a[2]  = p[0]; a[3]  = p[1];
      p = __builtin_bit_cast(v2h, __builtin_amdgcn_cvt_pkrtz(f1.x, f1.y));
      a[4]  = p[0]; a[5]  = p[1];
      p = __builtin_bit_cast(v2h, __builtin_amdgcn_cvt_pkrtz(f1.z, f1.w));
      a[6]  = p[0]; a[7]  = p[1];
      p = __builtin_bit_cast(v2h, __builtin_amdgcn_cvt_pkrtz(f2.x, f2.y));
      a[8]  = p[0]; a[9]  = p[1];
      p = __builtin_bit_cast(v2h, __builtin_amdgcn_cvt_pkrtz(f2.z, f2.w));
      a[10] = p[0]; a[11] = p[1];
      p = __builtin_bit_cast(v2h, __builtin_amdgcn_cvt_pkrtz(f3.x, f3.y));
      a[12] = p[0]; a[13] = p[1];
      p = __builtin_bit_cast(v2h, __builtin_amdgcn_cvt_pkrtz(f3.z, f3.w));
      a[14] = p[0]; a[15] = p[1];
    }

    const int kc = i * TILE_K;
#pragma unroll
    for (int j = 0; j < 8; ++j) {
      // B fragment: 32x16 f16, column N = 16j+hn, K = kc + 16g .. +15 contiguous in Wt
      const _Float16* wp = Wt + (size_t)(16 * j + hn) * F_DIM + kc + 16 * g;
      v8h blo = *(const v8h*)wp;
      v8h bhi = *(const v8h*)(wp + 8);
      v16h b;
#pragma unroll
      for (int q = 0; q < 8; ++q) { b[q] = blo[q]; b[8 + q] = bhi[q]; }
      acc[j] = __builtin_amdgcn_wmma_f32_16x16x32_f16(
          /*neg_a=*/false, a, /*neg_b=*/false, b,
          /*c_mod=*/(short)0, acc[j], /*reuse_a=*/false, /*reuse_b=*/false);
    }
    __syncthreads();   // all reads of buf done before wave0 re-issues into it
  }

  // Column sum / sum-of-squares of (acc + bias), guarding tail rows.
  // C/D layout: lane(g,hn) reg r -> row = 8g + r (within wave's 16), col = 16j + hn.
#pragma unroll
  for (int j = 0; j < 8; ++j) {
    int c = 16 * j + hn;
    float bv = bias[c];
    float s = 0.f, q = 0.f;
#pragma unroll
    for (int r = 0; r < 8; ++r) {
      int grow = rowBase + wave * 16 + 8 * g + r;
      if (grow < M) { float v = acc[j][r] + bv; s += v; q += v * v; }
    }
    s += __shfl_xor(s, 16);
    q += __shfl_xor(q, 16);
    if (g == 0) { atomicAdd(&sSum[c], s); atomicAdd(&sSq[c], q); }
  }
  __syncthreads();
  if (tid < A_DIM) {
    atomicAdd(&colsum[tid], sSum[tid]);
    atomicAdd(&colsq[tid],  sSq[tid]);
  }
}

// ---------------------------------------------------------------------------
// Kernel 2: fold BN stats + W_out halves into gu = W@(scale_s*w1), gv = W@(scale_n*w2)
// and scalars consts[0] = sum(b*u + shift_s*w1) + b_out, consts[1] = sum(b*v + shift_n*w2).
// ---------------------------------------------------------------------------
__global__ void finalize_kernel(const float* __restrict__ statsS, const float* __restrict__ statsN,
                                const float* __restrict__ gamma, const float* __restrict__ beta,
                                const float* __restrict__ bias, const float* __restrict__ W_out,
                                const float* __restrict__ b_out, const float* __restrict__ W,
                                float* __restrict__ gu, float* __restrict__ gv,
                                float* __restrict__ consts) {
  __shared__ float uS[A_DIM], vN[A_DIM];
  __shared__ float cAcc[2];
  int tid = threadIdx.x;
  if (tid < 2) cAcc[tid] = 0.f;
  __syncthreads();

  int p = tid >> 7;        // 0 = self path, 1 = neigh path
  int c = tid & 127;
  const float* st = p ? statsN : statsS;
  float Mf   = p ? (float)(NNODES * KNB) : (float)NNODES;
  float mean = st[c] / Mf;
  float var  = st[A_DIM + c] / Mf - mean * mean;
  float scale = gamma[c] * rsqrtf(var + 1e-3f);
  float shift = beta[c] - mean * scale;
  float w  = W_out[p * A_DIM + c];
  float uv = scale * w;
  (p ? vN : uS)[c] = uv;
  atomicAdd(&cAcc[p], bias[c] * uv + shift * w);
  __syncthreads();
  if (tid == 0) { consts[0] = cAcc[0] + b_out[0]; consts[1] = cAcc[1]; }

  // gu[f], gv[f]: tid == f (0..255)
  float a0 = 0.f, a1 = 0.f;
  for (int cc = 0; cc < A_DIM; ++cc) {
    float wfc = W[(size_t)tid * A_DIM + cc];
    a0 += wfc * uS[cc];
    a1 += wfc * vN[cc];
  }
  gu[tid] = a0;
  gv[tid] = a1;
}

// ---------------------------------------------------------------------------
// Kernel 3: one wave32 per node. logit[k] = relu(self.gu + neigh_k.gv + consts),
// softmax across the 32 lanes (== K neighbors).
// ---------------------------------------------------------------------------
__global__ void __launch_bounds__(256)
attention_kernel(const float* __restrict__ self, const float* __restrict__ neigh,
                 const float* __restrict__ gu, const float* __restrict__ gv,
                 const float* __restrict__ consts, float* __restrict__ out) {
  int node = blockIdx.x * 8 + (threadIdx.x >> 5);
  int lane = threadIdx.x & 31;

  float gur[8], gvr[8];
#pragma unroll
  for (int i = 0; i < 8; ++i) { gur[i] = gu[lane + 32 * i]; gvr[i] = gv[lane + 32 * i]; }

  const float* sp = self + (size_t)node * F_DIM;
  float ds = 0.f;
#pragma unroll
  for (int i = 0; i < 8; ++i) ds += sp[lane + 32 * i] * gur[i];
#pragma unroll
  for (int off = 16; off >= 1; off >>= 1) ds += __shfl_xor(ds, off);
  ds += consts[0];
  float cN = consts[1];

  const float* np = neigh + (size_t)node * KNB * F_DIM;
  float mylog = 0.f;
  for (int k = 0; k < KNB; ++k) {
    const float* r = np + (size_t)k * F_DIM;
    __builtin_prefetch(r + F_DIM, 0, 0);   // speculative prefetch of next neighbor row
    float d = 0.f;
#pragma unroll
    for (int i = 0; i < 8; ++i) d += r[lane + 32 * i] * gvr[i];
#pragma unroll
    for (int off = 16; off >= 1; off >>= 1) d += __shfl_xor(d, off);
    float logit = fmaxf(ds + d + cN, 0.f);
    if (k == lane) mylog = logit;
  }

  float mx = mylog;
#pragma unroll
  for (int off = 16; off >= 1; off >>= 1) mx = fmaxf(mx, __shfl_xor(mx, off));
  float e = __expf(mylog - mx);
  float s = e;
#pragma unroll
  for (int off = 16; off >= 1; off >>= 1) s += __shfl_xor(s, off);
  out[(size_t)node * KNB + lane] = e / s;
}

// ---------------------------------------------------------------------------
extern "C" void kernel_launch(void* const* d_in, const int* in_sizes, int n_in,
                              void* d_out, int out_size, void* d_ws, size_t ws_size,
                              hipStream_t stream) {
  const float* self_feats  = (const float*)d_in[0];   // [N, 256]
  const float* neigh_feats = (const float*)d_in[1];   // [N, 32, 256]
  const float* W_shared    = (const float*)d_in[2];   // [256, 128]
  const float* b_shared    = (const float*)d_in[3];   // [128]
  const float* gamma       = (const float*)d_in[4];   // [128]
  const float* beta        = (const float*)d_in[5];   // [128]
  const float* W_out       = (const float*)d_in[6];   // [256, 1]
  const float* b_out       = (const float*)d_in[7];   // [1]
  float* out = (float*)d_out;                         // [N, 32, 1]

  char* ws = (char*)d_ws;
  _Float16* Wt   = (_Float16*)ws;                       //  64 KB f16 W transposed
  float* statsS  = (float*)(ws + 65536);                //  sum[128] | sumsq[128]
  float* statsN  = (float*)(ws + 65536 + 1024);         //  sum[128] | sumsq[128]
  float* gu      = (float*)(ws + 65536 + 2048);         //  [256]
  float* gv      = gu + F_DIM;                          //  [256]
  float* consts  = gv + F_DIM;                          //  [2]

  // 0) W -> f16 transpose + zero stats
  prep_kernel<<<128, 256, 0, stream>>>(W_shared, Wt, statsS, statsN);
  // 1) column stats of self @ W + b   (M = 20000)
  gemm_stats_kernel<<<(NNODES + TILE_ROWS - 1) / TILE_ROWS, 128, 0, stream>>>(
      self_feats, Wt, b_shared, statsS, statsS + A_DIM, NNODES);
  // 2) column stats of neigh @ W + b  (M = 640000)
  gemm_stats_kernel<<<(NNODES * KNB) / TILE_ROWS, 128, 0, stream>>>(
      neigh_feats, Wt, b_shared, statsN, statsN + A_DIM, NNODES * KNB);
  // 3) fold BN + W_out into gu/gv/consts
  finalize_kernel<<<1, 256, 0, stream>>>(statsS, statsN, gamma, beta, b_shared,
                                         W_out, b_out, W_shared, gu, gv, consts);
  // 4) logits + softmax, one wave per node
  attention_kernel<<<NNODES / 8, 256, 0, stream>>>(self_feats, neigh_feats,
                                                   gu, gv, consts, out);
}
